// NetGAT_37220186588038
// MI455X (gfx1250) — matrix operand order repaired and checked by reading, not compiled, not run on hardware
//
#include <hip/hip_runtime.h>
#include <hip/hip_bf16.h>
#include <math.h>

typedef __attribute__((ext_vector_type(16))) __bf16 v16bf;
typedef __attribute__((ext_vector_type(8)))  __bf16 v8bf;
typedef __attribute__((ext_vector_type(8)))  float  v8f;

// ---------------- device helpers ----------------

__device__ __forceinline__ unsigned short f32_to_bf16(float f) {
  unsigned u = __float_as_uint(f);
  u += 0x7FFFu + ((u >> 16) & 1u);          // round-to-nearest-even
  return (unsigned short)(u >> 16);
}

__device__ __forceinline__ void atomicMaxF(float* addr, float val) {
  // monotonic-bit trick: works for all non-NaN floats (init = -inf)
  if (val >= 0.0f) atomicMax((int*)addr, __float_as_int(val));
  else             atomicMin((unsigned int*)addr, __float_as_uint(val));
}

// ---------------- prep kernels ----------------

__global__ void k_fill_f32(float* p, float v, int n) {
  int t = blockIdx.x * blockDim.x + threadIdx.x;
  if (t < n) p[t] = v;
}

__global__ void k_copy_f32(const float* __restrict__ s, float* __restrict__ d, int n) {
  int t = blockIdx.x * blockDim.x + threadIdx.x;
  if (t < n) d[t] = s[t];
}

__global__ void k_conv_bf16(const float* __restrict__ s, unsigned short* __restrict__ d, int n) {
  int t = blockIdx.x * blockDim.x + threadIdx.x;
  if (t < n) d[t] = f32_to_bf16(s[t]);
}

// W is [K, Nw] row-major fp32 (x @ W). Produce Bt [Nw, Kpad] bf16, zero-padded in K.
__global__ void k_transpose_w(const float* __restrict__ W, int K, int Nw,
                              unsigned short* __restrict__ Bt, int Kpad) {
  int t = blockIdx.x * blockDim.x + threadIdx.x;
  if (t >= Nw * Kpad) return;
  int nIdx = t / Kpad;
  int k    = t - nIdx * Kpad;
  Bt[t] = (k < K) ? f32_to_bf16(W[(size_t)k * Nw + nIdx]) : (unsigned short)0;
}

// ---------------- WMMA GEMM: C[M,Nout] = A[M,K](bf16) x Bt[Nout,K]^T (bf16) + bias ----------------
// A row-major with leading dim lda; Bt holds B transposed (column n contiguous over K), ldb = Kpad.
// One wave per 16x16 output tile. K % 32 == 0, M % 16 == 0, Nout % 16 == 0.

__global__ void k_wmma_gemm_bias(const __bf16* __restrict__ A, int lda,
                                 const __bf16* __restrict__ Bt, int ldb,
                                 const float* __restrict__ bias,
                                 float* __restrict__ C, int ldc,
                                 int M, int K, int Nout) {
  const int wave  = (blockIdx.x * blockDim.x + threadIdx.x) >> 5;  // wave32
  const int lane  = threadIdx.x & 31;
  const int tilesN = Nout >> 4;
  const int tileM  = wave / tilesN;
  const int tileN  = wave - tileM * tilesN;
  if (tileM * 16 >= M) return;                 // wave-uniform; EXEC stays all-ones past here

  const int m16 = lane & 15;
  const int g   = lane >> 4;
  const int arow = tileM * 16 + m16;
  const int bcol = tileN * 16 + m16;
  const __bf16* aptr = A  + (size_t)arow * lda;
  const __bf16* bptr = Bt + (size_t)bcol * ldb;

  v8f acc = {};
  for (int k0 = 0; k0 < K; k0 += 32) {
    // A 16x32 bf16 fragment: lane group g holds K = k0+8g..+7 and k0+16+8g..+7 (ISA layout)
    v8bf a0 = *(const v8bf*)(aptr + k0 + 8 * g);
    v8bf a1 = *(const v8bf*)(aptr + k0 + 16 + 8 * g);
    // B 32x16 bf16 fragment: lane group g holds K = k0+8g..+7 and k0+16+8g..+7 (ISA layout)
    v8bf b0 = *(const v8bf*)(bptr + k0 + 8 * g);
    v8bf b1 = *(const v8bf*)(bptr + k0 + 16 + 8 * g);
    v16bf av, bv;
#pragma unroll
    for (int j = 0; j < 8; ++j) {
      av[j] = a0[j]; av[j + 8] = a1[j];
      bv[j] = b0[j]; bv[j + 8] = b1[j];
    }
    acc = __builtin_amdgcn_wmma_f32_16x16x32_bf16(
        /*neg_a=*/false, av, /*neg_b=*/false, bv,
        /*c_mod=*/(short)0, acc, /*reuse_a=*/false, /*reuse_b=*/false);
  }

  const float bval = bias ? bias[bcol] : 0.0f;
#pragma unroll
  for (int r = 0; r < 8; ++r) {
    const int row = tileM * 16 + r + 8 * g;    // C layout: VGPR r -> M=r (lanes 0-15), M=r+8 (lanes 16-31)
    C[(size_t)row * ldc + bcol] = acc[r] + bval;
  }
}

// ---------------- edge-attention kernels ----------------
// qkvs layout per node (stride 256): q [0,64) | k [64,128) | v [128,192) | s [192,256)

__global__ void k_attn_pass1(const float* __restrict__ qkvs,
                             const int* __restrict__ src, const int* __restrict__ dst,
                             float* __restrict__ alphaBuf, float* __restrict__ amax,
                             int H, int C, float scale, int E) {
  int t = blockIdx.x * blockDim.x + threadIdx.x;
  if (t >= E * H) return;
  int e  = t / H;
  int hd = t - e * H;
  int s = src[e], d = dst[e];
  const float* q = qkvs + (size_t)d * 256 + hd * C;
  const float* k = qkvs + (size_t)s * 256 + 64 + hd * C;
  float a = 0.0f;
  for (int c = 0; c < C; c += 4) {
    float4 qa = *(const float4*)(q + c);
    float4 ka = *(const float4*)(k + c);
    a += qa.x * ka.x + qa.y * ka.y + qa.z * ka.z + qa.w * ka.w;
  }
  a *= scale;
  alphaBuf[t] = a;
  atomicMaxF(&amax[(size_t)d * H + hd], a);
}

__global__ void k_attn_pass2(const float* __restrict__ qkvs,
                             const int* __restrict__ src, const int* __restrict__ dst,
                             const float* __restrict__ alphaBuf, const float* __restrict__ amax,
                             float* __restrict__ denom, float* __restrict__ acc,
                             int H, int C, int E) {
  int t = blockIdx.x * blockDim.x + threadIdx.x;
  if (t >= E * H) return;
  int e  = t / H;
  int hd = t - e * H;
  int s = src[e], d = dst[e];
  float ex = __expf(alphaBuf[t] - amax[(size_t)d * H + hd]);
  atomicAdd(&denom[(size_t)d * H + hd], ex);
  const float* v = qkvs + (size_t)s * 256 + 128 + hd * C;
  float* ap = acc + (size_t)d * 64 + hd * C;
  for (int c = 0; c < C; ++c) atomicAdd(ap + c, ex * v[c]);
}

__global__ void k_combine(const float* __restrict__ acc, const float* __restrict__ denom,
                          const float* __restrict__ qkvs, float* __restrict__ hout,
                          int H, int C, int N) {
  int t = blockIdx.x * blockDim.x + threadIdx.x;
  if (t >= N * 64) return;
  int n = t >> 6;
  int d = t & 63;
  int hd = d / C;
  hout[t] = acc[t] / (denom[(size_t)n * H + hd] + 1e-16f)
          + qkvs[(size_t)n * 256 + 192 + d];
}

// Tanhshrink of m[N,48] into bf16 buffer [N,64] zero-padded (so readout GEMM2 can use K=64)
__global__ void k_tanhshrink_pad(const float* __restrict__ m, unsigned short* __restrict__ out, int N) {
  int t = blockIdx.x * blockDim.x + threadIdx.x;
  if (t >= N * 64) return;
  int n = t >> 6;
  int j = t & 63;
  float v = 0.0f;
  if (j < 48) {
    float x = m[(size_t)n * 48 + j];
    v = x - tanhf(x);
  }
  out[t] = f32_to_bf16(v);
}

// ---------------- host ----------------

static inline dim3 g1(long n) { return dim3((unsigned)((n + 255) / 256)); }

extern "C" void kernel_launch(void* const* d_in, const int* in_sizes, int n_in,
                              void* d_out, int out_size, void* d_ws, size_t ws_size,
                              hipStream_t stream) {
  const int N = in_sizes[0] / 64;        // 50000
  const int E = in_sizes[1] / 2;         // 500000
  const int NLAYERS = 7;                 // conv8 output is unused by the reference

  const float* x  = (const float*)d_in[0];
  const int*   ei = (const int*)d_in[1];
  const int* src = ei;
  const int* dst = ei + E;

  // --- detect flattening order of the nested params dict ---
  // insertion order per layer: Wq,bq,Wk,bk,Wv,bv,Ws,bs  -> in_sizes[3] == 64
  // sorted-key order per layer: Wk,Wq,Ws,Wv,bk,bq,bs,bv -> in_sizes[3] == 4096
  const bool sortedLayers = (in_sizes[3] == 4096);
  const float *Wq[8], *Wk[8], *Wv[8], *Ws[8], *bq[8], *bk[8], *bv[8], *bs[8];
  for (int l = 0; l < 8; ++l) {
    int b = 2 + 8 * l;
    if (sortedLayers) {
      Wk[l] = (const float*)d_in[b + 0]; Wq[l] = (const float*)d_in[b + 1];
      Ws[l] = (const float*)d_in[b + 2]; Wv[l] = (const float*)d_in[b + 3];
      bk[l] = (const float*)d_in[b + 4]; bq[l] = (const float*)d_in[b + 5];
      bs[l] = (const float*)d_in[b + 6]; bv[l] = (const float*)d_in[b + 7];
    } else {
      Wq[l] = (const float*)d_in[b + 0]; bq[l] = (const float*)d_in[b + 1];
      Wk[l] = (const float*)d_in[b + 2]; bk[l] = (const float*)d_in[b + 3];
      Wv[l] = (const float*)d_in[b + 4]; bv[l] = (const float*)d_in[b + 5];
      Ws[l] = (const float*)d_in[b + 6]; bs[l] = (const float*)d_in[b + 7];
    }
  }
  int rb = 2 + 64;
  const float *W1, *W2, *b1, *b2;
  if (in_sizes[rb + 1] == 1536) {        // sorted: W1,W2,b1,b2
    W1 = (const float*)d_in[rb + 0]; W2 = (const float*)d_in[rb + 1];
    b1 = (const float*)d_in[rb + 2]; b2 = (const float*)d_in[rb + 3];
  } else {                               // insertion: W1,b1,W2,b2
    W1 = (const float*)d_in[rb + 0]; b1 = (const float*)d_in[rb + 1];
    W2 = (const float*)d_in[rb + 2]; b2 = (const float*)d_in[rb + 3];
  }

  // --- workspace carve-up ---
  char* ws = (char*)d_ws;
  size_t o = 0;
  auto take = [&](size_t bytes) { size_t r = o; o += (bytes + 255) & ~(size_t)255; return r; };
  float*          hA    = (float*)(ws + take((size_t)N * 64 * 4));
  float*          hB    = (float*)(ws + take((size_t)N * 64 * 4));
  unsigned short* hbf   = (unsigned short*)(ws + take((size_t)N * 64 * 2));  // also mbf (readout)
  float*          qkvs  = (float*)(ws + take((size_t)N * 256 * 4));
  float*          alphaB= (float*)(ws + take((size_t)E * 4 * 4));
  float*          amax  = (float*)(ws + take((size_t)N * 4 * 4));
  float*          denom = (float*)(ws + take((size_t)N * 4 * 4));
  float*          acc   = (float*)(ws + take((size_t)N * 64 * 4));           // also mbuf (N*48 f32)
  unsigned short* wt    = (unsigned short*)(ws + take((size_t)NLAYERS * 256 * 64 * 2));
  unsigned short* w1t   = (unsigned short*)(ws + take((size_t)48 * 64 * 2));
  unsigned short* w2t   = (unsigned short*)(ws + take((size_t)32 * 64 * 2));
  float*          bcat  = (float*)(ws + take((size_t)NLAYERS * 256 * 4));
  if (o > ws_size) return;  // workspace too small; nothing sane to do

  // --- prep: transpose+convert weights to bf16, concat biases (q|k|v|s) ---
  for (int l = 0; l < NLAYERS; ++l) {
    unsigned short* wtl = wt + (size_t)l * 256 * 64;
    k_transpose_w<<<g1(64 * 64), 256, 0, stream>>>(Wq[l], 64, 64, wtl + 0 * 64 * 64, 64);
    k_transpose_w<<<g1(64 * 64), 256, 0, stream>>>(Wk[l], 64, 64, wtl + 1 * 64 * 64, 64);
    k_transpose_w<<<g1(64 * 64), 256, 0, stream>>>(Wv[l], 64, 64, wtl + 2 * 64 * 64, 64);
    k_transpose_w<<<g1(64 * 64), 256, 0, stream>>>(Ws[l], 64, 64, wtl + 3 * 64 * 64, 64);
    float* bl = bcat + (size_t)l * 256;
    k_copy_f32<<<1, 64, 0, stream>>>(bq[l], bl + 0,   64);
    k_copy_f32<<<1, 64, 0, stream>>>(bk[l], bl + 64,  64);
    k_copy_f32<<<1, 64, 0, stream>>>(bv[l], bl + 128, 64);
    k_copy_f32<<<1, 64, 0, stream>>>(bs[l], bl + 192, 64);
  }
  k_transpose_w<<<g1(48 * 64), 256, 0, stream>>>(W1, 64, 48, w1t, 64);
  k_transpose_w<<<g1(32 * 64), 256, 0, stream>>>(W2, 48, 32, w2t, 64);

  // --- 7 transformer-conv layers ---
  const float* h_cur = x;
  float* h_next = hA;
  for (int l = 0; l < NLAYERS; ++l) {
    const int H = (l == 0) ? 4 : 1;
    const int C = 64 / H;
    const float scale = (l == 0) ? 0.25f : 0.125f;   // 1/sqrt(C)

    k_conv_bf16<<<g1((long)N * 64), 256, 0, stream>>>(h_cur, hbf, N * 64);

    {   // fused q|k|v|s GEMM: [N,64] x [64,256]
      long waves = (long)(N / 16) * (256 / 16);
      k_wmma_gemm_bias<<<g1(waves * 32), 256, 0, stream>>>(
          (const __bf16*)hbf, 64, (const __bf16*)(wt + (size_t)l * 256 * 64), 64,
          bcat + (size_t)l * 256, qkvs, 256, N, 64, 256);
    }

    k_fill_f32<<<g1((long)N * H), 256, 0, stream>>>(amax, -INFINITY, N * H);
    k_fill_f32<<<g1((long)N * H), 256, 0, stream>>>(denom, 0.0f, N * H);
    k_fill_f32<<<g1((long)N * 64), 256, 0, stream>>>(acc, 0.0f, N * 64);

    k_attn_pass1<<<g1((long)E * H), 256, 0, stream>>>(qkvs, src, dst, alphaB, amax, H, C, scale, E);
    k_attn_pass2<<<g1((long)E * H), 256, 0, stream>>>(qkvs, src, dst, alphaB, amax, denom, acc, H, C, E);
    k_combine  <<<g1((long)N * 64), 256, 0, stream>>>(acc, denom, qkvs, h_next, H, C, N);

    h_cur  = h_next;
    h_next = (h_next == hA) ? hB : hA;
  }

  // --- readout: [N,64]x[64,48] -> tanhshrink -> [N,64(pad)]x[64,32] ---
  float* mbuf = acc;                       // reuse (N*48 f32 fits)
  unsigned short* mbf = hbf;               // reuse
  k_conv_bf16<<<g1((long)N * 64), 256, 0, stream>>>(h_cur, hbf, N * 64);
  {
    long waves = (long)(N / 16) * (48 / 16);
    k_wmma_gemm_bias<<<g1(waves * 32), 256, 0, stream>>>(
        (const __bf16*)hbf, 64, (const __bf16*)w1t, 64, b1, mbuf, 48, N, 64, 48);
  }
  k_tanhshrink_pad<<<g1((long)N * 64), 256, 0, stream>>>(mbuf, mbf, N);
  {
    long waves = (long)(N / 16) * (32 / 16);
    k_wmma_gemm_bias<<<g1(waves * 32), 256, 0, stream>>>(
        (const __bf16*)mbf, 64, (const __bf16*)w2t, 64, b2, (float*)d_out, 32, N, 64, 32);
  }
}
